// GATNet_38697655336973
// MI455X (gfx1250) — compile-verified, matrix operand order checked by reading, not run on hardware
//
#include <hip/hip_runtime.h>

// ---------------------------------------------------------------------------
// 2-layer GAT (PyG GATConv semantics) for gfx1250 (MI455X, wave32, WMMA).
// GEMMs: v_wmma_f32_16x16x32_f16; B double-buffered in LDS via async
// global->LDS (ASYNCcnt), overlapping DMA with matrix ops.
// Edge softmax: 3 passes (logits+atomic-max, exp+atomic-sum, weighted scatter).
// ---------------------------------------------------------------------------

#define USE_ASYNC_LDS 1

typedef __attribute__((ext_vector_type(16))) _Float16 v16h;
typedef __attribute__((ext_vector_type(8)))  float    v8f;

#define NNODES 50000
#define NEDGES 800000
#define NEDGEP 850000   // edges + self loops
#define KDIM   256
#define NHEADS 8

// ---- ordered-uint encoding of float for atomic max (monotonic, total order)
__device__ __forceinline__ unsigned encf(float f) {
  unsigned u = __float_as_uint(f);
  return (u & 0x80000000u) ? ~u : (u | 0x80000000u);
}
__device__ __forceinline__ float decf(unsigned u) {
  return (u & 0x80000000u) ? __uint_as_float(u & 0x7fffffffu)
                           : __uint_as_float(~u);
}

// ---------------------------------------------------------------------------
__global__ void k_f32_to_f16(const float* __restrict__ in,
                             _Float16* __restrict__ out, int n) {
  int i = blockIdx.x * blockDim.x + threadIdx.x;
  if (i < n) out[i] = (_Float16)in[i];
}

// W[k, col] row-major f32 -> Wt[col*K + k] col-major f16
__global__ void k_w_transpose_f16(const float* __restrict__ in,
                                  _Float16* __restrict__ out, int K, int NC) {
  int t = blockIdx.x * blockDim.x + threadIdx.x;
  if (t < K * NC) {
    int col = t / K, k = t - col * K;
    out[t] = (_Float16)in[k * NC + col];
  }
}

__global__ void k_zero_f32(float* __restrict__ p, int n) {
  int i = blockIdx.x * blockDim.x + threadIdx.x;
  if (i < n) p[i] = 0.0f;
}
__global__ void k_zero_u32(unsigned* __restrict__ p, int n) {
  int i = blockIdx.x * blockDim.x + threadIdx.x;
  if (i < n) p[i] = 0u;   // encodes "very negative" in ordered-uint space
}

// ---------------------------------------------------------------------------
// GEMM: C[M, NCT*16] = A[M,256](f16 row-major) x B(f16 col-major: B[col*256+k])
// Workgroup = 8 waves = 128 rows.  B K-slices are double-buffered in LDS via
// async global->LDS: slice k+1 is in flight while WMMAs consume slice k.
// One s_wait_asynccnt + barrier per K-step covers both publish and WAR reuse.
// Tail waves recompute tile 0 (store-predicated) so EXEC is all-ones for
// every WMMA as the ISA requires.
// A fragment (contiguous k order, B uses the same): lane L holds 16 contiguous f16 of row
// (L&15), K-half (L>>4)*16 -> one 32B global vector load.
// LDS column stride 40 f16 (80B): bank spread + 16B alignment for ds_load_b128.
// C layout: VGPR p -> row = p + 8*(lane>>4), col = lane&15.
// ---------------------------------------------------------------------------
template <int NCT>
__global__ __launch_bounds__(256) void k_gemm_f16(
    const _Float16* __restrict__ A, const _Float16* __restrict__ B,
    float* __restrict__ C, int M) {
  constexpr int BSTR = 40;                      // f16 per LDS column (32+8 pad)
  __shared__ _Float16 Bs[2][NCT * 16 * BSTR];

  const int K    = KDIM;
  const int tid  = threadIdx.x;
  const int wave = tid >> 5;
  const int lane = tid & 31;

  long row0   = ((long)blockIdx.x * 8 + wave) * 16;
  bool active = row0 < M;                       // wave-uniform (store gate only)
  long rowc   = active ? row0 : 0;              // tail waves redo tile 0

  const int r  = lane & 15;
  const int kh = (lane >> 4) * 16;

  v8f c[NCT];
#pragma unroll
  for (int j = 0; j < NCT; ++j) c[j] = (v8f){};

  const _Float16* ap = A + (size_t)(rowc + r) * K + kh;

  // ---- stage one B K-slice (NCT*16 cols x 64B) into Bs[buf] ---------------
  auto stage = [&](int buf, int k0) {
#pragma unroll
    for (int it = 0; it < (NCT * 64) / 256; ++it) {
      int i    = tid + it * 256;
      int col  = i >> 2;
      int part = i & 3;
      const _Float16* g = B + (size_t)col * KDIM + k0 + part * 8;
      _Float16*       l = &Bs[buf][col * BSTR + part * 8];
#if USE_ASYNC_LDS
      unsigned           loff = (unsigned)(unsigned long long)(void*)l;
      unsigned long long ga   = (unsigned long long)g;
      asm volatile("global_load_async_to_lds_b128 %0, %1, off"
                   :: "v"(loff), "v"(ga) : "memory");
#else
      *(uint4*)l = *(const uint4*)g;
#endif
    }
  };

  stage(0, 0);
#if USE_ASYNC_LDS
  asm volatile("s_wait_asynccnt 0x0" ::: "memory");
#endif
  __syncthreads();

  constexpr int KSTEPS = KDIM / 32;
  for (int ks = 0; ks < KSTEPS; ++ks) {
    const int buf = ks & 1;
    if (ks + 1 < KSTEPS) stage(buf ^ 1, (ks + 1) * 32);   // async, no wait yet

    v16h a = *(const v16h*)(ap + ks * 32);
#pragma unroll
    for (int j = 0; j < NCT; ++j) {
      v16h b = *(const v16h*)(&Bs[buf][(j * 16 + r) * BSTR + kh]);
      c[j] = __builtin_amdgcn_wmma_f32_16x16x32_f16(
          false, a, false, b, (short)0, c[j], false, false);
    }

#if USE_ASYNC_LDS
    asm volatile("s_wait_asynccnt 0x0" ::: "memory");
#endif
    __syncthreads();   // publish next slice; readers of this slice done (WAR)
  }

  if (active) {
    const int NC = NCT * 16;
    int crow = (int)row0 + (lane >> 4) * 8;
    int ccol = lane & 15;
#pragma unroll
    for (int j = 0; j < NCT; ++j) {
      float* cp = C + (size_t)crow * NC + j * 16 + ccol;
#pragma unroll
      for (int p = 0; p < 8; ++p) cp[(size_t)p * NC] = c[j][p];
    }
  }
}

// ---------------------------------------------------------------------------
// a_src[n,h] = sum_c h[n,h,c]*att_src[h,c];  a_dst likewise.
// ---------------------------------------------------------------------------
__global__ void k_attn_dots(const float* __restrict__ h,
                            const float* __restrict__ att_s,
                            const float* __restrict__ att_d,
                            float* __restrict__ as_, float* __restrict__ ad_,
                            int C) {
  int t = blockIdx.x * blockDim.x + threadIdx.x;
  if (t >= NNODES * NHEADS) return;
  int n = t >> 3, hd = t & 7;
  const float* hp  = h + (size_t)n * NHEADS * C + hd * C;
  const float* asp = att_s + hd * C;
  const float* adp = att_d + hd * C;
  float s0 = 0.f, s1 = 0.f;
  for (int c = 0; c < C; ++c) {
    float v = hp[c];
    s0 += v * asp[c];
    s1 += v * adp[c];
  }
  as_[t] = s0;
  ad_[t] = s1;
}

__device__ __forceinline__ void edge_sd(const int* __restrict__ ei, int e,
                                        int& s, int& d) {
  if (e < NEDGES) { s = ei[e]; d = ei[NEDGES + e]; }
  else            { s = e - NEDGES; d = s; }
}

// pass 1: e = leaky_relu(a_src[s]+a_dst[d]); store; atomic-max per (dst,head)
__global__ void k_edge_logits(const int* __restrict__ ei,
                              const float* __restrict__ as_,
                              const float* __restrict__ ad_,
                              float* __restrict__ ev,
                              unsigned* __restrict__ m) {
  int e = blockIdx.x * blockDim.x + threadIdx.x;
  if (e >= NEDGEP) return;
  int s, d;
  edge_sd(ei, e, s, d);
  float4 a0 = *(const float4*)(as_ + (size_t)s * 8);
  float4 a1 = *(const float4*)(as_ + (size_t)s * 8 + 4);
  float4 b0 = *(const float4*)(ad_ + (size_t)d * 8);
  float4 b1 = *(const float4*)(ad_ + (size_t)d * 8 + 4);
  float v[8] = {a0.x + b0.x, a0.y + b0.y, a0.z + b0.z, a0.w + b0.w,
                a1.x + b1.x, a1.y + b1.y, a1.z + b1.z, a1.w + b1.w};
#pragma unroll
  for (int hd = 0; hd < NHEADS; ++hd) {
    v[hd] = v[hd] > 0.f ? v[hd] : 0.2f * v[hd];       // leaky relu, slope 0.2
    atomicMax(&m[d * NHEADS + hd], encf(v[hd]));
  }
  *(float4*)(ev + (size_t)e * 8)     = make_float4(v[0], v[1], v[2], v[3]);
  *(float4*)(ev + (size_t)e * 8 + 4) = make_float4(v[4], v[5], v[6], v[7]);
}

// pass 2: w = exp(e - max[dst]); store; atomic-add into denom[dst]
__global__ void k_edge_exp(const int* __restrict__ ei,
                           float* __restrict__ ev,
                           const unsigned* __restrict__ m,
                           float* __restrict__ denom) {
  int e = blockIdx.x * blockDim.x + threadIdx.x;
  if (e >= NEDGEP) return;
  int s, d;
  edge_sd(ei, e, s, d);
  float4 e0 = *(const float4*)(ev + (size_t)e * 8);
  float4 e1 = *(const float4*)(ev + (size_t)e * 8 + 4);
  uint4  m0 = *(const uint4*)(m + (size_t)d * 8);
  uint4  m1 = *(const uint4*)(m + (size_t)d * 8 + 4);
  float w[8];
  w[0] = __expf(e0.x - decf(m0.x));
  w[1] = __expf(e0.y - decf(m0.y));
  w[2] = __expf(e0.z - decf(m0.z));
  w[3] = __expf(e0.w - decf(m0.w));
  w[4] = __expf(e1.x - decf(m1.x));
  w[5] = __expf(e1.y - decf(m1.y));
  w[6] = __expf(e1.z - decf(m1.z));
  w[7] = __expf(e1.w - decf(m1.w));
#pragma unroll
  for (int hd = 0; hd < NHEADS; ++hd)
    __hip_atomic_fetch_add(&denom[d * NHEADS + hd], w[hd], __ATOMIC_RELAXED,
                           __HIP_MEMORY_SCOPE_AGENT);
  *(float4*)(ev + (size_t)e * 8)     = make_float4(w[0], w[1], w[2], w[3]);
  *(float4*)(ev + (size_t)e * 8 + 4) = make_float4(w[4], w[5], w[6], w[7]);
}

// pass 3: out[dst] += h[src] * alpha. One wave32 per edge; each lane moves a
// float4 (head index is uniform inside each 4-aligned chunk since HW | 16).
template <int CH, int HW>
__global__ __launch_bounds__(256) void k_edge_msg(
    const int* __restrict__ ei, const float* __restrict__ hsrc,
    const float* __restrict__ ev, const float* __restrict__ denom,
    float* __restrict__ out) {
  int wave = (blockIdx.x * blockDim.x + threadIdx.x) >> 5;
  int lane = threadIdx.x & 31;
  if (wave >= NEDGEP) return;
  int s, d;
  edge_sd(ei, wave, s, d);
  const float* hp = hsrc + (size_t)s * CH;
  float*       op = out + (size_t)d * CH;
#pragma unroll
  for (int t = 0; t < CH / 128; ++t) {
    int j  = lane * 4 + t * 128;
    int hd = j / HW;
    float alpha = ev[(size_t)wave * NHEADS + hd] /
                  (denom[d * NHEADS + hd] + 1e-16f);
    float4 v = *(const float4*)(hp + j);
    __hip_atomic_fetch_add(op + j + 0, v.x * alpha, __ATOMIC_RELAXED,
                           __HIP_MEMORY_SCOPE_AGENT);
    __hip_atomic_fetch_add(op + j + 1, v.y * alpha, __ATOMIC_RELAXED,
                           __HIP_MEMORY_SCOPE_AGENT);
    __hip_atomic_fetch_add(op + j + 2, v.z * alpha, __ATOMIC_RELAXED,
                           __HIP_MEMORY_SCOPE_AGENT);
    __hip_atomic_fetch_add(op + j + 3, v.w * alpha, __ATOMIC_RELAXED,
                           __HIP_MEMORY_SCOPE_AGENT);
  }
}

// hin16 = f16(elu(out1 + bias1))   (row width 256 = power of two)
__global__ void k_elu_f16(const float* __restrict__ in,
                          const float* __restrict__ bias,
                          _Float16* __restrict__ out, int n) {
  int i = blockIdx.x * blockDim.x + threadIdx.x;
  if (i < n) {
    float v = in[i] + bias[i & 255];
    v = v > 0.f ? v : (__expf(v) - 1.0f);
    out[i] = (_Float16)v;
  }
}

// out[n,c] = mean_h out2[n,h,c] + bias2[c]
__global__ void k_head_mean(const float* __restrict__ out2,
                            const float* __restrict__ bias,
                            float* __restrict__ out) {
  int t = blockIdx.x * blockDim.x + threadIdx.x;
  if (t >= NNODES * 16) return;
  int n = t >> 4, c = t & 15;
  float s = 0.f;
#pragma unroll
  for (int hd = 0; hd < NHEADS; ++hd) s += out2[(size_t)n * 128 + hd * 16 + c];
  out[t] = s * 0.125f + bias[c];
}

// ---------------------------------------------------------------------------
static inline size_t alignup(size_t x) { return (x + 255) & ~(size_t)255; }

extern "C" void kernel_launch(void* const* d_in, const int* in_sizes, int n_in,
                              void* d_out, int out_size, void* d_ws,
                              size_t ws_size, hipStream_t stream) {
  (void)in_sizes; (void)n_in; (void)out_size; (void)ws_size;

  const float* x        = (const float*)d_in[0];
  const int*   ei       = (const int*)d_in[1];
  const float* W1       = (const float*)d_in[2];
  const float* att_src1 = (const float*)d_in[3];
  const float* att_dst1 = (const float*)d_in[4];
  const float* bias1    = (const float*)d_in[5];
  const float* W2       = (const float*)d_in[6];
  const float* att_src2 = (const float*)d_in[7];
  const float* att_dst2 = (const float*)d_in[8];
  const float* bias2    = (const float*)d_in[9];
  float*       out      = (float*)d_out;

  // ---- workspace carving (~162 MB, with reuse) ----------------------------
  char* p = (char*)d_ws;
  auto take = [&](size_t bytes) { char* q = p; p += alignup(bytes); return q; };

  _Float16* xh   = (_Float16*)take((size_t)NNODES * KDIM * 2); // x f16; reused for elu(h1) f16
  _Float16* Wt1  = (_Float16*)take((size_t)KDIM * 256 * 2);
  _Float16* Wt2  = (_Float16*)take((size_t)KDIM * 128 * 2);
  float*    h1   = (float*)take((size_t)NNODES * 256 * 4);     // later: h2 | out2
  float*    out1 = (float*)take((size_t)NNODES * 256 * 4);
  float*    asrc = (float*)take((size_t)NNODES * NHEADS * 4);
  float*    adst = (float*)take((size_t)NNODES * NHEADS * 4);
  float*    ev   = (float*)take((size_t)NEDGEP * NHEADS * 4);
  unsigned* mbuf = (unsigned*)take((size_t)NNODES * NHEADS * 4);
  float*    dden = (float*)take((size_t)NNODES * NHEADS * 4);

  float* h2   = h1;                     // reuse (h1 dead after layer-1 pass 3)
  float* out2 = h1 + (size_t)NNODES * 128;

  const int T = 256;
  auto blk = [](long n, int t) { return dim3((unsigned)((n + t - 1) / t)); };
  const unsigned gemmBlocks = (unsigned)((NNODES / 16 + 7) / 8);   // 8 waves/block

  // ==== layer 1 ============================================================
  k_f32_to_f16<<<blk((long)NNODES * KDIM, T), T, 0, stream>>>(x, xh, NNODES * KDIM);
  k_w_transpose_f16<<<blk(256 * 256, T), T, 0, stream>>>(W1, Wt1, KDIM, 256);
  k_w_transpose_f16<<<blk(256 * 128, T), T, 0, stream>>>(W2, Wt2, KDIM, 128);

  // h1 = x @ W1
  k_gemm_f16<16><<<gemmBlocks, T, 0, stream>>>(xh, Wt1, h1, NNODES);

  k_attn_dots<<<blk((long)NNODES * NHEADS, T), T, 0, stream>>>(h1, att_src1, att_dst1, asrc, adst, 32);

  k_zero_u32<<<blk((long)NNODES * NHEADS, T), T, 0, stream>>>(mbuf, NNODES * NHEADS);
  k_zero_f32<<<blk((long)NNODES * NHEADS, T), T, 0, stream>>>(dden, NNODES * NHEADS);
  k_zero_f32<<<blk((long)NNODES * 256, T), T, 0, stream>>>(out1, NNODES * 256);

  k_edge_logits<<<blk(NEDGEP, T), T, 0, stream>>>(ei, asrc, adst, ev, mbuf);
  k_edge_exp<<<blk(NEDGEP, T), T, 0, stream>>>(ei, ev, mbuf, dden);
  k_edge_msg<256, 32><<<blk((long)NEDGEP * 32, T), T, 0, stream>>>(ei, h1, ev, dden, out1);

  // hin16 = f16(elu(out1 + bias1))   (reuses xh buffer)
  k_elu_f16<<<blk((long)NNODES * 256, T), T, 0, stream>>>(out1, bias1, xh, NNODES * 256);

  // ==== layer 2 ============================================================
  k_gemm_f16<8><<<gemmBlocks, T, 0, stream>>>(xh, Wt2, h2, NNODES);

  k_attn_dots<<<blk((long)NNODES * NHEADS, T), T, 0, stream>>>(h2, att_src2, att_dst2, asrc, adst, 16);

  k_zero_u32<<<blk((long)NNODES * NHEADS, T), T, 0, stream>>>(mbuf, NNODES * NHEADS);
  k_zero_f32<<<blk((long)NNODES * NHEADS, T), T, 0, stream>>>(dden, NNODES * NHEADS);
  k_zero_f32<<<blk((long)NNODES * 128, T), T, 0, stream>>>(out2, NNODES * 128);

  k_edge_logits<<<blk(NEDGEP, T), T, 0, stream>>>(ei, asrc, adst, ev, mbuf);
  k_edge_exp<<<blk(NEDGEP, T), T, 0, stream>>>(ei, ev, mbuf, dden);
  k_edge_msg<128, 16><<<blk((long)NEDGEP * 32, T), T, 0, stream>>>(ei, h2, ev, dden, out2);

  k_head_mean<<<blk((long)NNODES * 16, T), T, 0, stream>>>(out2, bias2, out);
}
